// CausalSelfAttention_28192165331551
// MI455X (gfx1250) — matrix operand order repaired and checked by reading, not compiled, not run on hardware
//
#include <hip/hip_runtime.h>
#include <stdint.h>

// Problem constants (from reference)
#define BQ   2
#define SQ   2048
#define HIDQ 2048
#define NHQ  16
#define HSQ  128
#define ROTQ 32
#define MROWS (BQ * SQ)        // 4096 flattened (b,s) rows
#define NQKV  (3 * HIDQ)       // 6144 qkv output columns

typedef float  v8f   __attribute__((ext_vector_type(8)));
typedef __bf16 v16bf __attribute__((ext_vector_type(16)));

union Frag16 {
    v16bf v;
    unsigned int u[8];
    unsigned short h[16];
};

__device__ __forceinline__ unsigned short f2bf(float f) {
    unsigned int u = __float_as_uint(f);
    unsigned int r = u + 0x7FFFu + ((u >> 16) & 1u);   // round-to-nearest-even
    return (unsigned short)(r >> 16);
}

__device__ __forceinline__ v8f vzero8() {
    v8f z;
    #pragma unroll
    for (int i = 0; i < 8; ++i) z[i] = 0.0f;
    return z;
}

// A-matrix 16x32 bf16 pair offsets: VGPR r holds K = {2r (r<4) | 8+2r (r>=4)} + 8*khalf, +1
__device__ __forceinline__ int a_pk(int r, int kh) { return (r < 4 ? 2 * r : 8 + 2 * r) + 8 * kh; }
// B-matrix 32x16 bf16 pair offsets: VGPR r holds K = {2r (r<4) | 8+2r (r>=4)} + 8*khalf, +1
__device__ __forceinline__ int b_pk(int r, int kh) { return (r < 4 ? 2 * r : 8 + 2 * r) + 8 * kh; }

// ------------------------------------------------- Tensor Data Mover (TDM) --
#if defined(__AMDGCN__) && __has_builtin(__builtin_amdgcn_tensor_load_to_lds) && \
    __has_builtin(__builtin_amdgcn_s_wait_tensorcnt)
#define USE_TDM 1
#endif

#ifdef USE_TDM
typedef unsigned int v4u  __attribute__((ext_vector_type(4)));
typedef int          v8it __attribute__((ext_vector_type(8)));
typedef int          v4it __attribute__((ext_vector_type(4)));

// 2D tile DMA: global (row-major, bf16, row stride = stride0 elems) -> LDS
// (contiguous, rows of tile_d0). Descriptor per CDNA5 ISA:
//  g0: [1:0]=count=1 | [63:32]=lds_addr | [120:64]=global_addr | [127:126]=type=2
//  g1: [17:16]=data_size=1 (2B) | [79:48]=tensor_dim0 | [111:80]=tensor_dim1
//      | [127:112]=tile_dim0 | [143:128]=tile_dim1 | [207:160]=tensor_dim0_stride
__device__ __forceinline__ void tdm_load_tile_2d(unsigned lds_off, const void* gptr,
                                                 unsigned tile_d0, unsigned tile_d1,
                                                 unsigned stride0) {
    unsigned long long ga = (unsigned long long)(uintptr_t)gptr;
    v4u g0;
    g0[0] = 1u;                                                  // count=1
    g0[1] = lds_off;                                             // lds_addr
    g0[2] = (unsigned)ga;                                        // global_addr[31:0]
    g0[3] = (unsigned)((ga >> 32) & 0x01FFFFFFu) | (2u << 30);   // addr[56:32] | type=2
    v8it g1;
    g1[0] = (int)(1u << 16);                                     // data_size=2B, mask=0
    g1[1] = (int)((tile_d0 & 0xFFFFu) << 16);                    // tensor_dim0[15:0]
    g1[2] = (int)((tile_d0 >> 16) | ((tile_d1 & 0xFFFFu) << 16));// dim0[31:16]|dim1[15:0]
    g1[3] = (int)((tile_d1 >> 16) | ((tile_d0 & 0xFFFFu) << 16));// dim1[31:16]|tile_dim0
    g1[4] = (int)(tile_d1 & 0xFFFFu);                            // tile_dim1, tile_dim2=0
    g1[5] = (int)stride0;                                        // dim0_stride[31:0]
    g1[6] = 0;
    g1[7] = 0;
    v4it z4 = {0, 0, 0, 0};
#if __has_include(<hip/amd_detail/amd_gfx1250_TDM.h>)
    v8it z8 = {0, 0, 0, 0, 0, 0, 0, 0};
    __builtin_amdgcn_tensor_load_to_lds(g0, g1, z4, z4, z8, 0);  // 6-arg toolchain
#else
    __builtin_amdgcn_tensor_load_to_lds(g0, g1, z4, z4, 0);      // 5-arg ROCm 7.2
#endif
}
#endif

// ---------------------------------------------------------------- converts --
__global__ void cvt_bf16(const float* __restrict__ src, unsigned short* __restrict__ dst, int n) {
    int i = blockIdx.x * blockDim.x + threadIdx.x;
    if (i < n) dst[i] = f2bf(src[i]);
}

// src [K][N] f32 -> dst [N][K] bf16 (weight transpose so GEMM is TN / K-contiguous)
__global__ void transpose_bf16(const float* __restrict__ src, unsigned short* __restrict__ dst,
                               int K, int N) {
    long i = (long)blockIdx.x * blockDim.x + threadIdx.x;
    long total = (long)K * N;
    if (i < total) {
        int  k = (int)(i % K);
        long n = i / K;
        dst[n * (long)K + k] = f2bf(src[(long)k * N + n]);
    }
}

// ------------------------------------------------------- QKV GEMM + RoPE ----
__global__ void __launch_bounds__(256) qkv_gemm_rope(
    const unsigned short* __restrict__ xb,
    const unsigned short* __restrict__ WqkvT,
    const float* __restrict__ bqkv,
    unsigned short* __restrict__ Qb,
    unsigned short* __restrict__ Kb,
    unsigned short* __restrict__ VtB)
{
    __shared__ unsigned short As[128][32];
    __shared__ unsigned short Bs[128][32];

    const int t = threadIdx.x;
    const int wave = t >> 5, lane = t & 31;
    const int wm = wave & 3, wn = wave >> 2;          // 4 waves in M, 2 in N
    const int ln = lane & 15, kh = lane >> 4;
    const int tileM = blockIdx.y;                     // 32 tiles of 128 rows
    const int tileN = blockIdx.x;                     // 48 tiles of 128 cols
    const int lrow = t >> 1, half = t & 1;

    v8f acc[2][4];
    #pragma unroll
    for (int mi = 0; mi < 2; ++mi)
        #pragma unroll
        for (int ni = 0; ni < 4; ++ni) acc[mi][ni] = vzero8();

    const unsigned short* aRow = xb    + (size_t)(tileM * 128 + lrow) * HIDQ + half * 16;
    const unsigned short* bRow = WqkvT + (size_t)(tileN * 128 + lrow) * HIDQ + half * 16;

    for (int k0 = 0; k0 < HIDQ; k0 += 32) {
        __syncthreads();
        {   // A tile 128x32 / B tile 128x32 (rows are output cols, K-contiguous)
            const uint4* pa = reinterpret_cast<const uint4*>(aRow + k0);
            *reinterpret_cast<uint4*>(&As[lrow][half * 16])     = pa[0];
            *reinterpret_cast<uint4*>(&As[lrow][half * 16 + 8]) = pa[1];
            const uint4* pb = reinterpret_cast<const uint4*>(bRow + k0);
            *reinterpret_cast<uint4*>(&Bs[lrow][half * 16])     = pb[0];
            *reinterpret_cast<uint4*>(&Bs[lrow][half * 16 + 8]) = pb[1];
        }
        if (k0 + 32 < HIDQ) {   // prefetch next K-step (global_prefetch_b8)
            __builtin_prefetch(aRow + k0 + 32, 0, 1);
            __builtin_prefetch(bRow + k0 + 32, 0, 1);
        }
        __syncthreads();

        Frag16 bf[4];
        #pragma unroll
        for (int ni = 0; ni < 4; ++ni) {
            const unsigned short* bp = &Bs[wn * 64 + ni * 16 + ln][0];
            #pragma unroll
            for (int r = 0; r < 8; ++r)
                bf[ni].u[r] = *reinterpret_cast<const unsigned int*>(bp + b_pk(r, kh));
        }
        #pragma unroll
        for (int mi = 0; mi < 2; ++mi) {
            Frag16 af;
            const unsigned short* ap = &As[wm * 32 + mi * 16 + ln][0];
            #pragma unroll
            for (int r = 0; r < 8; ++r)
                af.u[r] = *reinterpret_cast<const unsigned int*>(ap + a_pk(r, kh));
            #pragma unroll
            for (int ni = 0; ni < 4; ++ni)
                acc[mi][ni] = __builtin_amdgcn_wmma_f32_16x16x32_bf16(
                    false, af.v, false, bf[ni].v, (short)0, acc[mi][ni], false, false);
        }
    }

    // ---- epilogue: bias + RoPE + scatter ----
    const int h = tileN / 3, blk = tileN % 3;         // blk: 0=q 1=k 2=v

    #pragma unroll
    for (int mi = 0; mi < 2; ++mi)
        #pragma unroll
        for (int ni = 0; ni < 4; ++ni) {
            float bias = bqkv[tileN * 128 + wn * 64 + ni * 16 + ln];
            #pragma unroll
            for (int r = 0; r < 8; ++r) acc[mi][ni][r] += bias;
        }

    if (blk < 2 && wn == 0) {
        // RoPE: cols d = ln (acc ni=0) pair with d+16 (acc ni=1); freq index j = d % 16
        float invf = expf(-(float)ln * 0.57564627324851147f);   // 10000^(-ln/16)
        #pragma unroll
        for (int mi = 0; mi < 2; ++mi)
            #pragma unroll
            for (int r = 0; r < 8; ++r) {
                int mg = tileM * 128 + wm * 32 + mi * 16 + r + 8 * kh;
                int s  = mg & (SQ - 1);
                float ang = (float)s * invf;
                float c = cosf(ang), sn = sinf(ang);
                float a0 = acc[mi][0][r], a1 = acc[mi][1][r];
                acc[mi][0][r] = a0 * c - a1 * sn;
                acc[mi][1][r] = a1 * c + a0 * sn;
            }
    }

    if (blk < 2) {
        unsigned short* dst = (blk == 0) ? Qb : Kb;
        #pragma unroll
        for (int mi = 0; mi < 2; ++mi)
            #pragma unroll
            for (int ni = 0; ni < 4; ++ni) {
                int d = wn * 64 + ni * 16 + ln;
                #pragma unroll
                for (int r = 0; r < 8; ++r) {
                    int mg = tileM * 128 + wm * 32 + mi * 16 + r + 8 * kh;
                    int b  = mg >> 11, s = mg & (SQ - 1);
                    size_t bh = (size_t)(b * NHQ + h);
                    dst[(bh * SQ + s) * HSQ + d] = f2bf(acc[mi][ni][r]);
                }
            }
    } else {
        // V^T: elements r, r+1 are consecutive sequence positions -> packed b32 stores
        #pragma unroll
        for (int mi = 0; mi < 2; ++mi)
            #pragma unroll
            for (int ni = 0; ni < 4; ++ni) {
                int d = wn * 64 + ni * 16 + ln;
                #pragma unroll
                for (int r = 0; r < 8; r += 2) {
                    int mg = tileM * 128 + wm * 32 + mi * 16 + r + 8 * kh;
                    int b  = mg >> 11, s = mg & (SQ - 1);
                    size_t bh = (size_t)(b * NHQ + h);
                    unsigned lo = f2bf(acc[mi][ni][r]);
                    unsigned hi = f2bf(acc[mi][ni][r + 1]);
                    *reinterpret_cast<unsigned int*>(&VtB[(bh * HSQ + d) * SQ + s]) =
                        lo | (hi << 16);
                }
            }
    }
}

// --------------------------------------------------- flash attention --------
// One block = (b,h) x 128 query rows; 8 waves x 16 q-rows each.
__global__ void __launch_bounds__(256) attn_kernel(
    const unsigned short* __restrict__ Qb,
    const unsigned short* __restrict__ Kb,
    const unsigned short* __restrict__ VtB,
    unsigned short* __restrict__ attnb)     // [B][S][HID] bf16
{
    __shared__ unsigned short Ks[32][128];   // 32 keys x 128 d
    __shared__ unsigned short Vs[128][32];   // 128 d x 32 keys (V^T)
    __shared__ unsigned short Ps[8][16][32]; // per-wave P bounce

    const int t = threadIdx.x, wave = t >> 5, lane = t & 31;
    const int ln = lane & 15, kh = lane >> 4;
    const int tileQ = blockIdx.x;            // 16 tiles of 128 q rows
    const int bh    = blockIdx.y;            // 32 (b,h) pairs
    const int qbase = tileQ * 128 + wave * 16;
    const int lrow = t >> 1, half = t & 1;

    // preload Q fragments (A-layout, 4 d-steps of 32)
    Frag16 qf[4];
    {
        const unsigned short* qp = Qb + ((size_t)bh * SQ + qbase + ln) * HSQ;
        #pragma unroll
        for (int ds = 0; ds < 4; ++ds)
            #pragma unroll
            for (int r = 0; r < 8; ++r)
                qf[ds].u[r] = *reinterpret_cast<const unsigned int*>(qp + ds * 32 + a_pk(r, kh));
    }

    v8f o[8];
    #pragma unroll
    for (int i = 0; i < 8; ++i) o[i] = vzero8();
    float rmax[8], rsum[8];
    #pragma unroll
    for (int r = 0; r < 8; ++r) { rmax[r] = -1e30f; rsum[r] = 0.0f; }

    const float scale = 0.08838834764831845f;  // 1/sqrt(128)
    const int kend = (tileQ + 1) * 128;

    for (int k0 = 0; k0 < kend; k0 += 32) {
        __syncthreads();
#ifdef USE_TDM
        // Tensor Data Mover stages both tiles; wave 0 issues, waits TENSORcnt.
        if (wave == 0) {
            tdm_load_tile_2d((unsigned)(uintptr_t)&Ks[0][0],
                             Kb + ((size_t)bh * SQ + k0) * HSQ, HSQ, 32, HSQ);
            tdm_load_tile_2d((unsigned)(uintptr_t)&Vs[0][0],
                             VtB + (size_t)bh * HSQ * SQ + k0, 32, HSQ, SQ);
            __builtin_amdgcn_s_wait_tensorcnt(0);
        }
#else
        {   // manual staging fallback
            int key = t >> 3, seg = t & 7;
            const uint4* p = reinterpret_cast<const uint4*>(
                Kb + ((size_t)bh * SQ + k0 + key) * HSQ + seg * 16);
            *reinterpret_cast<uint4*>(&Ks[key][seg * 16])     = p[0];
            *reinterpret_cast<uint4*>(&Ks[key][seg * 16 + 8]) = p[1];
            const uint4* pv = reinterpret_cast<const uint4*>(
                VtB + ((size_t)bh * HSQ + lrow) * SQ + k0 + half * 16);
            *reinterpret_cast<uint4*>(&Vs[lrow][half * 16])     = pv[0];
            *reinterpret_cast<uint4*>(&Vs[lrow][half * 16 + 8]) = pv[1];
        }
#endif
        if (k0 + 32 < kend) {   // prefetch next k-tile
            __builtin_prefetch(Kb + ((size_t)bh * SQ + k0 + 32 + (t >> 3)) * HSQ + (t & 7) * 16, 0, 1);
            __builtin_prefetch(VtB + ((size_t)bh * HSQ + lrow) * SQ + k0 + 32 + half * 16, 0, 1);
        }
        __syncthreads();

        // ---- scores S = Q K^T (16 q x 32 k) ----
        v8f sacc[2];
        sacc[0] = vzero8(); sacc[1] = vzero8();
        #pragma unroll
        for (int ds = 0; ds < 4; ++ds)
            #pragma unroll
            for (int nsub = 0; nsub < 2; ++nsub) {
                Frag16 bfk;
                const unsigned short* kp = &Ks[nsub * 16 + ln][ds * 32];
                #pragma unroll
                for (int r = 0; r < 8; ++r)
                    bfk.u[r] = *reinterpret_cast<const unsigned int*>(kp + b_pk(r, kh));
                sacc[nsub] = __builtin_amdgcn_wmma_f32_16x16x32_bf16(
                    false, qf[ds].v, false, bfk.v, (short)0, sacc[nsub], false, false);
            }

        // ---- causal mask + scale + online softmax ----
        #pragma unroll
        for (int r = 0; r < 8; ++r) {
            int q = qbase + r + 8 * kh;
            #pragma unroll
            for (int nsub = 0; nsub < 2; ++nsub) {
                int kk = k0 + nsub * 16 + ln;
                sacc[nsub][r] = sacc[nsub][r] * scale + (kk > q ? -1e9f : 0.0f);
            }
            float v = fmaxf(sacc[0][r], sacc[1][r]);
            #pragma unroll
            for (int off = 1; off < 16; off <<= 1) v = fmaxf(v, __shfl_xor(v, off, 32));
            float nm  = fmaxf(rmax[r], v);
            float fac = expf(rmax[r] - nm);
            float p0  = expf(sacc[0][r] - nm);
            float p1  = expf(sacc[1][r] - nm);
            sacc[0][r] = p0; sacc[1][r] = p1;
            float ps = p0 + p1;
            #pragma unroll
            for (int off = 1; off < 16; off <<= 1) ps += __shfl_xor(ps, off, 32);
            rsum[r] = rsum[r] * fac + ps;
            rmax[r] = nm;
            #pragma unroll
            for (int i = 0; i < 8; ++i) o[i][r] *= fac;
        }

        // ---- P: C-layout -> A-layout via per-wave LDS bounce ----
        #pragma unroll
        for (int nsub = 0; nsub < 2; ++nsub)
            #pragma unroll
            for (int r = 0; r < 8; ++r)
                Ps[wave][r + 8 * kh][nsub * 16 + ln] = f2bf(sacc[nsub][r]);

        Frag16 pf;
        {
            const unsigned short* pp = &Ps[wave][ln][0];
            #pragma unroll
            for (int r = 0; r < 8; ++r)
                pf.u[r] = *reinterpret_cast<const unsigned int*>(pp + a_pk(r, kh));
        }

        // ---- O += P V ----
        #pragma unroll
        for (int dsub = 0; dsub < 8; ++dsub) {
            Frag16 vf;
            const unsigned short* vp = &Vs[dsub * 16 + ln][0];
            #pragma unroll
            for (int r = 0; r < 8; ++r)
                vf.u[r] = *reinterpret_cast<const unsigned int*>(vp + b_pk(r, kh));
            o[dsub] = __builtin_amdgcn_wmma_f32_16x16x32_bf16(
                false, pf.v, false, vf.v, (short)0, o[dsub], false, false);
        }
    }

    // ---- finalize ----
    const int b = bh / NHQ, h = bh % NHQ;
    #pragma unroll
    for (int dsub = 0; dsub < 8; ++dsub) {
        int d = dsub * 16 + ln;
        #pragma unroll
        for (int r = 0; r < 8; ++r) {
            int q = qbase + r + 8 * kh;
            float val = o[dsub][r] / rsum[r];
            attnb[((size_t)(b * SQ + q)) * HIDQ + h * HSQ + d] = f2bf(val);
        }
    }
}

// ------------------------------------------------------- dense output GEMM --
__global__ void __launch_bounds__(256) dense_gemm(
    const unsigned short* __restrict__ Ab,       // [4096][2048] bf16
    const unsigned short* __restrict__ BT,       // [2048][2048] bf16 (W^T)
    const float* __restrict__ bias,              // [2048]
    float* __restrict__ out)                     // [4096][2048] f32
{
    __shared__ unsigned short As[128][32];
    __shared__ unsigned short Bs[128][32];

    const int t = threadIdx.x;
    const int wave = t >> 5, lane = t & 31;
    const int wm = wave & 3, wn = wave >> 2;
    const int ln = lane & 15, kh = lane >> 4;
    const int tileM = blockIdx.y;
    const int tileN = blockIdx.x;
    const int lrow = t >> 1, half = t & 1;

    v8f acc[2][4];
    #pragma unroll
    for (int mi = 0; mi < 2; ++mi)
        #pragma unroll
        for (int ni = 0; ni < 4; ++ni) acc[mi][ni] = vzero8();

    const unsigned short* aRow = Ab + (size_t)(tileM * 128 + lrow) * HIDQ + half * 16;
    const unsigned short* bRow = BT + (size_t)(tileN * 128 + lrow) * HIDQ + half * 16;

    for (int k0 = 0; k0 < HIDQ; k0 += 32) {
        __syncthreads();
        {
            const uint4* pa = reinterpret_cast<const uint4*>(aRow + k0);
            *reinterpret_cast<uint4*>(&As[lrow][half * 16])     = pa[0];
            *reinterpret_cast<uint4*>(&As[lrow][half * 16 + 8]) = pa[1];
            const uint4* pb = reinterpret_cast<const uint4*>(bRow + k0);
            *reinterpret_cast<uint4*>(&Bs[lrow][half * 16])     = pb[0];
            *reinterpret_cast<uint4*>(&Bs[lrow][half * 16 + 8]) = pb[1];
        }
        if (k0 + 32 < HIDQ) {
            __builtin_prefetch(aRow + k0 + 32, 0, 1);
            __builtin_prefetch(bRow + k0 + 32, 0, 1);
        }
        __syncthreads();

        Frag16 bf[4];
        #pragma unroll
        for (int ni = 0; ni < 4; ++ni) {
            const unsigned short* bp = &Bs[wn * 64 + ni * 16 + ln][0];
            #pragma unroll
            for (int r = 0; r < 8; ++r)
                bf[ni].u[r] = *reinterpret_cast<const unsigned int*>(bp + b_pk(r, kh));
        }
        #pragma unroll
        for (int mi = 0; mi < 2; ++mi) {
            Frag16 af;
            const unsigned short* ap = &As[wm * 32 + mi * 16 + ln][0];
            #pragma unroll
            for (int r = 0; r < 8; ++r)
                af.u[r] = *reinterpret_cast<const unsigned int*>(ap + a_pk(r, kh));
            #pragma unroll
            for (int ni = 0; ni < 4; ++ni)
                acc[mi][ni] = __builtin_amdgcn_wmma_f32_16x16x32_bf16(
                    false, af.v, false, bf[ni].v, (short)0, acc[mi][ni], false, false);
        }
    }

    #pragma unroll
    for (int mi = 0; mi < 2; ++mi)
        #pragma unroll
        for (int ni = 0; ni < 4; ++ni) {
            int n = tileN * 128 + wn * 64 + ni * 16 + ln;
            float bv = bias[n];
            #pragma unroll
            for (int r = 0; r < 8; ++r) {
                int mg = tileM * 128 + wm * 32 + mi * 16 + r + 8 * kh;
                out[(size_t)mg * HIDQ + n] = acc[mi][ni][r] + bv;
            }
        }
}

// ---------------------------------------------------------------------------
extern "C" void kernel_launch(void* const* d_in, const int* in_sizes, int n_in,
                              void* d_out, int out_size, void* d_ws, size_t ws_size,
                              hipStream_t stream) {
    (void)in_sizes; (void)n_in; (void)out_size; (void)ws_size;
    const float* x      = (const float*)d_in[0];
    // d_in[1] position_ids (arange, computed inline), d_in[2] mask (causal, computed inline)
    const float* Wqkv   = (const float*)d_in[3];
    const float* bqkv   = (const float*)d_in[4];
    const float* Wdense = (const float*)d_in[5];
    const float* bdense = (const float*)d_in[6];
    float* out = (float*)d_out;

    char* ws = (char*)d_ws;
    size_t off = 0;
    auto alloc = [&](size_t bytes) -> void* {
        void* p = ws + off;
        off += (bytes + 255) & ~(size_t)255;
        return p;
    };
    unsigned short* xb      = (unsigned short*)alloc((size_t)MROWS * HIDQ * 2);
    unsigned short* WqkvT   = (unsigned short*)alloc((size_t)NQKV * HIDQ * 2);
    unsigned short* WdenseT = (unsigned short*)alloc((size_t)HIDQ * HIDQ * 2);
    unsigned short* Qb      = (unsigned short*)alloc((size_t)BQ * NHQ * SQ * HSQ * 2);
    unsigned short* Kb      = (unsigned short*)alloc((size_t)BQ * NHQ * SQ * HSQ * 2);
    unsigned short* Vt      = (unsigned short*)alloc((size_t)BQ * NHQ * SQ * HSQ * 2);
    unsigned short* attnb   = (unsigned short*)alloc((size_t)MROWS * HIDQ * 2);

    int n1 = MROWS * HIDQ;
    cvt_bf16<<<(n1 + 255) / 256, 256, 0, stream>>>(x, xb, n1);
    long n2 = (long)HIDQ * NQKV;
    transpose_bf16<<<(int)((n2 + 255) / 256), 256, 0, stream>>>(Wqkv, WqkvT, HIDQ, NQKV);
    long n3 = (long)HIDQ * HIDQ;
    transpose_bf16<<<(int)((n3 + 255) / 256), 256, 0, stream>>>(Wdense, WdenseT, HIDQ, HIDQ);

    qkv_gemm_rope<<<dim3(NQKV / 128, MROWS / 128), 256, 0, stream>>>(xb, WqkvT, bqkv, Qb, Kb, Vt);
    attn_kernel<<<dim3(SQ / 128, BQ * NHQ), 256, 0, stream>>>(Qb, Kb, Vt, attnb);
    dense_gemm<<<dim3(HIDQ / 128, MROWS / 128), 256, 0, stream>>>(attnb, WdenseT, bdense, out);
}
